// DART_54443005444356
// MI455X (gfx1250) — matrix operand order repaired and checked by reading, not compiled, not run on hardware
//
#include <hip/hip_runtime.h>
#include <hip/hip_bf16.h>
#include <math.h>

#define BB  2
#define SS  2048
#define HHH 768
#define NHH 12
#define HDD 64
#define III 3072
#define MMM (BB*SS)

typedef __attribute__((ext_vector_type(16))) _Float16 v16h;
typedef __attribute__((ext_vector_type(8)))  _Float16 v8h;
typedef __attribute__((ext_vector_type(8)))  float    v8f;

__device__ __forceinline__ v8f wmma_f16f32(v16h a, v16h b, v8f c) {
    // D = A(16x32 f16) * B(32x16 f16) + C(16x16 f32)
    return __builtin_amdgcn_wmma_f32_16x16x32_f16(
        /*neg_a=*/false, a, /*neg_b=*/false, b,
        /*c_mod=*/(short)0, c, /*reuse_a=*/false, /*reuse_b=*/false);
}

// A-fragment loader: row-major f16 row pointer -> 16x32 A layout
// lane<16 holds K {0..15}; lane>=16 holds K {16..31}
__device__ __forceinline__ v16h load_afrag(const _Float16* row, int k0, int hi) {
    v8h a0 = *(const v8h*)(row + k0 + hi * 16);
    v8h a1 = *(const v8h*)(row + k0 + hi * 16 + 8);
    return __builtin_shufflevector(a0, a1, 0,1,2,3,4,5,6,7,8,9,10,11,12,13,14,15);
}

// ---------------------------------------------------------------------------
// Weight prep: f32 (K x N) row-major  ->  f16 transposed (N x K) row-major.
// ---------------------------------------------------------------------------
__global__ void k_wconv(const float* __restrict__ w, _Float16* __restrict__ wT,
                        int K, int N) {
    int idx = blockIdx.x * blockDim.x + threadIdx.x;
    if (idx >= K * N) return;
    int k = idx / N, n = idx - k * N;
    wT[(size_t)n * K + k] = (_Float16)w[idx];
}

// ---------------------------------------------------------------------------
// AdaLN conditioning MLP (2 rows only -> scalar kernels, negligible cost)
// ---------------------------------------------------------------------------
__global__ void k_dp1(const float* __restrict__ e, const float* __restrict__ w,
                      const float* __restrict__ bias, float* __restrict__ out) {
    int idx = blockIdx.x * blockDim.x + threadIdx.x;
    if (idx >= BB * 3072) return;
    int bb = idx / 3072, o = idx - bb * 3072;
    float s = bias[o];
    const float* er = e + bb * HHH;
    for (int k = 0; k < HHH; ++k) s = fmaf(er[k], w[(size_t)k * 3072 + o], s);
    out[idx] = s / (1.f + __expf(-s));  // SiLU
}

__global__ void k_dp2(const float* __restrict__ a, const float* __restrict__ w,
                      const float* __restrict__ bias, float* __restrict__ out) {
    int idx = blockIdx.x * blockDim.x + threadIdx.x;
    if (idx >= BB * 1536) return;
    int bb = idx / 1536, o = idx - bb * 1536;
    float s = bias[o];
    const float* ar = a + bb * 3072;
    for (int k = 0; k < 3072; ++k) s = fmaf(ar[k], w[(size_t)k * 1536 + o], s);
    out[idx] = s;   // dp[b][0:768]=scale, dp[b][768:1536]=shift
}

// ---------------------------------------------------------------------------
// LayerNorm + (1+scale)*h + shift, emit f16 activations (one block per row)
// ---------------------------------------------------------------------------
__global__ void k_ln_mod(const float* __restrict__ x, const float* __restrict__ g,
                         const float* __restrict__ bta, const float* __restrict__ dp,
                         _Float16* __restrict__ out) {
    int row = blockIdx.x;          // 0..4095
    int bb  = row >> 11;           // row / 2048
    const float* xr = x + (size_t)row * HHH;
    __shared__ float s1[256], s2[256];
    float a0 = 0.f, a1 = 0.f, vals[3];
    #pragma unroll
    for (int i = 0; i < 3; ++i) {
        int c = threadIdx.x + i * 256;
        float v = xr[c];
        vals[i] = v; a0 += v; a1 += v * v;
    }
    s1[threadIdx.x] = a0; s2[threadIdx.x] = a1;
    __syncthreads();
    for (int off = 128; off > 0; off >>= 1) {
        if ((int)threadIdx.x < off) {
            s1[threadIdx.x] += s1[threadIdx.x + off];
            s2[threadIdx.x] += s2[threadIdx.x + off];
        }
        __syncthreads();
    }
    float mean = s1[0] * (1.f / HHH);
    float var  = s2[0] * (1.f / HHH) - mean * mean;
    float rstd = rsqrtf(var + 1e-12f);
    #pragma unroll
    for (int i = 0; i < 3; ++i) {
        int c = threadIdx.x + i * 256;
        float hn = (vals[i] - mean) * rstd * g[c] + bta[c];
        float sc = dp[bb * 1536 + c];
        float sh = dp[bb * 1536 + 768 + c];
        out[(size_t)row * HHH + c] = (_Float16)(hn * (1.f + sc) + sh);
    }
}

// ---------------------------------------------------------------------------
// WMMA GEMM: C(M x N) = A(M x K, f16) * BT(N x K, f16)^T + bias
// block = 128 thr = 4 waves; each wave computes a 32x64 tile (2 A-frags x
// 4 B-frags -> 8 WMMAs per K-step, B reused across both row fragments).
// grid (M/128, N/64).
// EPI 0: scatter f16 to (B,NH,S,HD)   EPI 1: f32 out + residual   EPI 2: GELU f16
// ---------------------------------------------------------------------------
template <int EPI>
__global__ void __launch_bounds__(128, 1)
k_gemm(const _Float16* __restrict__ A, const _Float16* __restrict__ BT,
       const float* __restrict__ bias, const float* __restrict__ res,
       void* __restrict__ outp, int N, int K) {
    int wave = threadIdx.x >> 5;
    int lane = threadIdx.x & 31;
    int l15 = lane & 15, hi = lane >> 4;
    int rowBase = blockIdx.x * 128 + wave * 32;
    int colBase = blockIdx.y * 64;

    v8f acc[2][4] = {};
    const _Float16* arow0 = A + (size_t)(rowBase + l15) * K;
    const _Float16* arow1 = arow0 + (size_t)16 * K;

    for (int k0 = 0; k0 < K; k0 += 32) {
        v16h af0 = load_afrag(arow0, k0, hi);
        v16h af1 = load_afrag(arow1, k0, hi);
        #pragma unroll
        for (int t = 0; t < 4; ++t) {
            // B-frag (32x16): lane = column, lo lanes K 0..15, hi lanes K 16..31
            const _Float16* bcol = BT + (size_t)(colBase + t * 16 + l15) * K;
            v16h bf = *(const v16h*)(bcol + k0 + hi * 16);
            acc[0][t] = wmma_f16f32(af0, bf, acc[0][t]);
            acc[1][t] = wmma_f16f32(af1, bf, acc[1][t]);
        }
    }

    #pragma unroll
    for (int rb = 0; rb < 2; ++rb) {
        #pragma unroll
        for (int t = 0; t < 4; ++t) {
            int col = colBase + t * 16 + l15;
            float bvv = bias[col];
            #pragma unroll
            for (int r = 0; r < 8; ++r) {
                int row = rowBase + rb * 16 + hi * 8 + r;
                float v = acc[rb][t][r] + bvv;
                if (EPI == 0) {          // QKV scatter: (b,s,h,d) -> (b,nh,s,hd) f16
                    int bbx = row >> 11, s = row & 2047;
                    int hh = col >> 6,  d = col & 63;
                    ((_Float16*)outp)[(((size_t)bbx * NHH + hh) * SS + s) * HDD + d] =
                        (_Float16)v;
                } else if (EPI == 1) {   // f32 + residual
                    ((float*)outp)[(size_t)row * N + col] = v + res[(size_t)row * N + col];
                } else {                 // exact GELU, f16 out
                    float gl = 0.5f * v * (1.f + erff(v * 0.70710678118f));
                    ((_Float16*)outp)[(size_t)row * N + col] = (_Float16)gl;
                }
            }
        }
    }
}

// ---------------------------------------------------------------------------
// Causal flash attention, HD=64. grid (S/64, B*NH), block 128 (4 waves x 16 q rows)
// QK^T: 2 WMMAs per 16x16 score tile (d contraction 64); PV: 1 WMMA per 16-col
// d-tile (key contraction 32). P transits D-layout -> A-layout via LDS.
// ---------------------------------------------------------------------------
__global__ void __launch_bounds__(128, 1)
k_attn(const _Float16* __restrict__ Q, const _Float16* __restrict__ K,
       const _Float16* __restrict__ V, _Float16* __restrict__ ctxOut) {
    int bh = blockIdx.y;
    int bb = bh / NHH, hh = bh - bb * NHH;
    int qBlk = blockIdx.x * 64;
    int wave = threadIdx.x >> 5, lane = threadIdx.x & 31;
    int l15 = lane & 15, hi = lane >> 4;
    int qRow0 = qBlk + wave * 16;

    __shared__ __align__(64) _Float16 Pbuf[4][16][32];  // wave-private P tiles
    __shared__ __align__(64) _Float16 VT[64][32];       // V block transposed (d x key)

    const _Float16* Kb = K + (size_t)bh * SS * HDD;
    const _Float16* Vb = V + (size_t)bh * SS * HDD;

    // Q fragments (two K-steps over d=0..63)
    const _Float16* qrow = Q + ((size_t)bh * SS + qRow0 + l15) * HDD;
    v16h qf0 = load_afrag(qrow, 0, hi);
    v16h qf1 = load_afrag(qrow, 32, hi);

    float mrow[8], lrow[8];
    #pragma unroll
    for (int r = 0; r < 8; ++r) { mrow[r] = -1e30f; lrow[r] = 0.f; }
    v8f cacc[4] = {};

    for (int j = 0; j < qBlk + 64; j += 32) {
        __syncthreads();
        {   // cooperative V block transpose: 32 keys x 64 d -> VT[d][key]
            int key = threadIdx.x >> 2;
            int dc  = (threadIdx.x & 3) * 16;
            const _Float16* vr = Vb + (size_t)(j + key) * HDD + dc;
            #pragma unroll
            for (int i = 0; i < 16; ++i) VT[dc + i][key] = vr[i];
        }
        // prefetch next K/V block while this one computes (global_prefetch_b8)
        if (j + 32 < qBlk + 64) {
            __builtin_prefetch(Kb + (size_t)(j + 32 + lane) * HDD, 0, 0);
            __builtin_prefetch(Vb + (size_t)(j + 32 + lane) * HDD, 0, 0);
        }
        __syncthreads();

        // scores: two 16x16 tiles (keys j..j+15, j+16..j+31)
        v8f sc[2] = {};
        #pragma unroll
        for (int tt = 0; tt < 2; ++tt) {
            const _Float16* kr = Kb + (size_t)(j + tt * 16 + l15) * HDD;
            v16h kfa = *(const v16h*)(kr + hi * 16);
            v16h kfb = *(const v16h*)(kr + 32 + hi * 16);
            sc[tt] = wmma_f16f32(qf0, kfa, sc[tt]);
            sc[tt] = wmma_f16f32(qf1, kfb, sc[tt]);
        }

        // online softmax (rows striped: vgpr r -> row hi*8+r, 16-lane xor reductions)
        float alpha[8];
        #pragma unroll
        for (int r = 0; r < 8; ++r) {
            int qr = qRow0 + hi * 8 + r;
            float s0 = sc[0][r] * 0.125f + ((j + l15)      > qr ? -10000.f : 0.f);
            float s1 = sc[1][r] * 0.125f + ((j + 16 + l15) > qr ? -10000.f : 0.f);
            float mx = fmaxf(s0, s1);
            #pragma unroll
            for (int mk = 1; mk < 16; mk <<= 1) mx = fmaxf(mx, __shfl_xor(mx, mk, 32));
            float mn = fmaxf(mrow[r], mx);
            float e0 = __expf(s0 - mn), e1 = __expf(s1 - mn);
            float rs = e0 + e1;
            #pragma unroll
            for (int mk = 1; mk < 16; mk <<= 1) rs += __shfl_xor(rs, mk, 32);
            float al = __expf(mrow[r] - mn);
            lrow[r] = lrow[r] * al + rs;
            mrow[r] = mn;
            alpha[r] = al;
            int pr = hi * 8 + r;
            Pbuf[wave][pr][l15]      = (_Float16)e0;
            Pbuf[wave][pr][16 + l15] = (_Float16)e1;
        }
        #pragma unroll
        for (int t = 0; t < 4; ++t)
            #pragma unroll
            for (int r = 0; r < 8; ++r) cacc[t][r] *= alpha[r];

        __threadfence_block();  // order wave-private LDS P writes before frag reads

        // P fragment (A-layout) and PV WMMAs (contraction over 32 keys)
        const _Float16* pr2 = &Pbuf[wave][l15][0];
        v8h p0 = *(const v8h*)(pr2 + hi * 16);
        v8h p1 = *(const v8h*)(pr2 + hi * 16 + 8);
        v16h pf = __builtin_shufflevector(p0, p1, 0,1,2,3,4,5,6,7,8,9,10,11,12,13,14,15);
        #pragma unroll
        for (int t = 0; t < 4; ++t) {
            v16h vf = *(const v16h*)(&VT[t * 16 + l15][hi * 16]);
            cacc[t] = wmma_f16f32(pf, vf, cacc[t]);
        }
    }

    // normalize and store ctx as f16 (B*S, H) for the O-projection GEMM
    #pragma unroll
    for (int t = 0; t < 4; ++t) {
        int d = t * 16 + l15;
        #pragma unroll
        for (int r = 0; r < 8; ++r) {
            int qr = qRow0 + hi * 8 + r;
            float o = cacc[t][r] / lrow[r];
            ctxOut[((size_t)bb * SS + qr) * HHH + hh * HDD + d] = (_Float16)o;
        }
    }
}

// ---------------------------------------------------------------------------
extern "C" void kernel_launch(void* const* d_in, const int* in_sizes, int n_in,
                              void* d_out, int out_size, void* d_ws, size_t ws_size,
                              hipStream_t stream) {
    (void)in_sizes; (void)n_in; (void)out_size; (void)ws_size;
    const float* x    = (const float*)d_in[0];
    const float* demb = (const float*)d_in[1];
    const float* ln1g = (const float*)d_in[2];
    const float* ln1b = (const float*)d_in[3];
    const float* ln2g = (const float*)d_in[4];
    const float* ln2b = (const float*)d_in[5];
    const float* wq = (const float*)d_in[6];   const float* bq = (const float*)d_in[7];
    const float* wk = (const float*)d_in[8];   const float* bk = (const float*)d_in[9];
    const float* wv = (const float*)d_in[10];  const float* bv = (const float*)d_in[11];
    const float* wo = (const float*)d_in[12];  const float* bo = (const float*)d_in[13];
    const float* wf1 = (const float*)d_in[14]; const float* bf1 = (const float*)d_in[15];
    const float* wf2 = (const float*)d_in[16]; const float* bf2 = (const float*)d_in[17];
    const float* wd1 = (const float*)d_in[18]; const float* bd1 = (const float*)d_in[19];
    const float* wd2 = (const float*)d_in[20]; const float* bd2 = (const float*)d_in[21];

    char* ws = (char*)d_ws;
    size_t off = 0;
    auto carve = [&](size_t bytes) -> char* {
        char* p = ws + off;
        off += (bytes + 255) & ~(size_t)255;
        return p;
    };
    _Float16* wqT  = (_Float16*)carve((size_t)768 * 768 * 2);
    _Float16* wkT  = (_Float16*)carve((size_t)768 * 768 * 2);
    _Float16* wvT  = (_Float16*)carve((size_t)768 * 768 * 2);
    _Float16* woT  = (_Float16*)carve((size_t)768 * 768 * 2);
    _Float16* wf1T = (_Float16*)carve((size_t)3072 * 768 * 2);
    _Float16* wf2T = (_Float16*)carve((size_t)768 * 3072 * 2);
    float*    dp1  = (float*)carve((size_t)BB * 3072 * 4);
    float*    dp2  = (float*)carve((size_t)BB * 1536 * 4);
    _Float16* h1   = (_Float16*)carve((size_t)MMM * HHH * 2);
    _Float16* qb   = (_Float16*)carve((size_t)MMM * HHH * 2);
    _Float16* kb   = (_Float16*)carve((size_t)MMM * HHH * 2);
    _Float16* vb   = (_Float16*)carve((size_t)MMM * HHH * 2);
    _Float16* ctx  = (_Float16*)carve((size_t)MMM * HHH * 2);
    float*    xmid = (float*)carve((size_t)MMM * HHH * 4);
    _Float16* h2   = (_Float16*)carve((size_t)MMM * HHH * 2);
    _Float16* act1 = (_Float16*)carve((size_t)MMM * III * 2);

    // 1) weight conversion (fp32 -> f16 transposed)
    k_wconv<<<(768 * 768 + 255) / 256, 256, 0, stream>>>(wq, wqT, 768, 768);
    k_wconv<<<(768 * 768 + 255) / 256, 256, 0, stream>>>(wk, wkT, 768, 768);
    k_wconv<<<(768 * 768 + 255) / 256, 256, 0, stream>>>(wv, wvT, 768, 768);
    k_wconv<<<(768 * 768 + 255) / 256, 256, 0, stream>>>(wo, woT, 768, 768);
    k_wconv<<<(768 * 3072 + 255) / 256, 256, 0, stream>>>(wf1, wf1T, 768, 3072);
    k_wconv<<<(3072 * 768 + 255) / 256, 256, 0, stream>>>(wf2, wf2T, 3072, 768);

    // 2) AdaLN conditioning
    k_dp1<<<(BB * 3072 + 255) / 256, 256, 0, stream>>>(demb, wd1, bd1, dp1);
    k_dp2<<<(BB * 1536 + 255) / 256, 256, 0, stream>>>(dp1, wd2, bd2, dp2);

    // 3) LN1 + modulate
    k_ln_mod<<<MMM, 256, 0, stream>>>(x, ln1g, ln1b, dp2, h1);

    // 4) QKV projections (WMMA)
    dim3 g1(MMM / 128, HHH / 64);
    k_gemm<0><<<g1, 128, 0, stream>>>(h1, wqT, bq, nullptr, qb, HHH, HHH);
    k_gemm<0><<<g1, 128, 0, stream>>>(h1, wkT, bk, nullptr, kb, HHH, HHH);
    k_gemm<0><<<g1, 128, 0, stream>>>(h1, wvT, bv, nullptr, vb, HHH, HHH);

    // 5) causal flash attention (WMMA)
    k_attn<<<dim3(SS / 64, BB * NHH), 128, 0, stream>>>(qb, kb, vb, ctx);

    // 6) O-projection + residual (f32)
    k_gemm<1><<<g1, 128, 0, stream>>>(ctx, woT, bo, x, xmid, HHH, HHH);

    // 7) LN2 + modulate
    k_ln_mod<<<MMM, 256, 0, stream>>>(xmid, ln2g, ln2b, dp2, h2);

    // 8) FFN: GEMM+GELU, then GEMM + residual -> d_out (f32)
    dim3 g2(MMM / 128, III / 64);
    k_gemm<2><<<g2, 128, 0, stream>>>(h2, wf1T, bf1, nullptr, act1, III, HHH);
    k_gemm<1><<<g1, 128, 0, stream>>>(act1, wf2T, bf2, xmid, d_out, HHH, III);
}
